// Prior_12730283065883
// MI455X (gfx1250) — compile-verified, matrix operand order checked by reading, not run on hardware
//
#include <hip/hip_runtime.h>
#include <hip/hip_bf16.h>
#include <math.h>

typedef __attribute__((ext_vector_type(16))) __bf16 v16bf;
typedef __attribute__((ext_vector_type(8)))  __bf16 v8bf;
typedef __attribute__((ext_vector_type(4)))  __bf16 v4bf;
typedef __attribute__((ext_vector_type(8)))  float  v8f;
typedef __attribute__((ext_vector_type(4)))  float  v4f;

constexpr int NB = 4096;   // batch
constexpr int NA = 32;     // action
constexpr int NS = 1024;   // sample
constexpr int NR = 2048;   // rnn hidden
constexpr int NH = 2048;   // mlp hidden
constexpr int NK0 = NA + NS;  // 1056, multiple of 32

// ---------------------------------------------------------------------------
// CDNA5 async global->LDS copies (ASYNCcnt-tracked, no VGPR round-trip).
// IOFFSET is added to BOTH the LDS and the global address (ISA 08_async_tensor
// §4.4), so one base pair covers the two 16B halves of a 32B row chunk.
// ---------------------------------------------------------------------------
__device__ __forceinline__ void async_cp16(uint32_t lds, const void* g) {
  asm volatile("global_load_async_to_lds_b128 %0, %1, off"
               :: "v"(lds), "v"(g) : "memory");
}
__device__ __forceinline__ void async_cp16_off16(uint32_t lds, const void* g) {
  asm volatile("global_load_async_to_lds_b128 %0, %1, off offset:16"
               :: "v"(lds), "v"(g) : "memory");
}
__device__ __forceinline__ uint32_t lds_off(const void* p) {
  // Flat pointers to LDS keep the LDS byte address in addr[31:0] (ISA §10.2).
  return (uint32_t)(uintptr_t)p;
}

// ---------------------------------------------------------------------------
// Elementwise helpers
// ---------------------------------------------------------------------------

__global__ void pack_xcat(const float* __restrict__ act, const float* __restrict__ ps,
                          __bf16* __restrict__ x) {
  int i = blockIdx.x * 256 + threadIdx.x;           // over NB*NK0
  if (i >= NB * NK0) return;
  int b = i / NK0, c = i - b * NK0;
  float v = (c < NA) ? act[b * NA + c] : ps[b * NS + (c - NA)];
  x[i] = (__bf16)v;
}

__global__ void cvt_f32_bf16x4(const float* __restrict__ s, __bf16* __restrict__ d, int n4) {
  int i = blockIdx.x * 256 + threadIdx.x;
  if (i >= n4) return;
  v4f v = ((const v4f*)s)[i];
  v4bf o;
  o[0] = (__bf16)v[0]; o[1] = (__bf16)v[1]; o[2] = (__bf16)v[2]; o[3] = (__bf16)v[3];
  ((v4bf*)d)[i] = o;
}

__global__ void gru_eltwise(const float* __restrict__ ig, const float* __restrict__ hg,
                            const float* __restrict__ prevh, const float* __restrict__ b_n,
                            float* __restrict__ nh_f, __bf16* __restrict__ nh_b) {
  int i = blockIdx.x * 256 + threadIdx.x;           // over NB*NR
  if (i >= NB * NR) return;
  int b = i >> 11;            // / 2048
  int j = i & (NR - 1);
  size_t base = (size_t)b * (3 * NR);
  float igr = ig[base + j], igz = ig[base + NR + j], ign = ig[base + 2 * NR + j];
  float hgr = hg[base + j], hgz = hg[base + NR + j], hgn = hg[base + 2 * NR + j];
  float reset  = 1.f / (1.f + expf(-(igr + hgr)));
  float update = 1.f / (1.f + expf(-(igz + hgz)));
  float nv = tanhf(ign + reset * (hgn + b_n[j]));
  float pv = prevh[i];
  float out = nv + update * (pv - nv);
  nh_f[i] = out;
  nh_b[i] = (__bf16)out;
}

__global__ void head_eltwise(const float* __restrict__ o, const float* __restrict__ eps,
                             float* __restrict__ mean, float* __restrict__ stdv,
                             float* __restrict__ samp) {
  int i = blockIdx.x * 256 + threadIdx.x;           // over NB*NS
  if (i >= NB * NS) return;
  int b = i >> 10, j = i & (NS - 1);
  float m  = o[(size_t)b * (2 * NS) + j];
  float sp = o[(size_t)b * (2 * NS) + NS + j];
  float st = ((sp > 20.f) ? sp : log1pf(expf(sp))) + 0.1f;
  mean[i] = m;
  stdv[i] = st;
  samp[i] = m + st * eps[i];
}

// ---------------------------------------------------------------------------
// Tiled WMMA GEMM:  Y[M,N] = act(X[M,K] * W[N,K]^T + bias[N])
// Block: 256 threads = 8 waves (2x4), tile 128x128, BK=32, double-buffered
// async global->LDS staging. Wave tile 64x32 => 4x2 grid of 16x16 WMMA accs.
// A/B fragments follow the CDNA5 16-bit A-matrix layout:
//   lane%16 -> row (M for A / N for B), lane/16 -> K half,
//   elems 0..7  = K[(lane>>4)*8 .. +7],  elems 8..15 = K[16+(lane>>4)*8 .. +7]
// ---------------------------------------------------------------------------

template <int ACT>  // 0 = identity, 1 = ELU
__global__ void __launch_bounds__(256)
gemm_bt(const __bf16* __restrict__ X, const __bf16* __restrict__ W,
        const float* __restrict__ bias,
        float* __restrict__ Cf, __bf16* __restrict__ Cb,
        int M, int N, int K) {
  __shared__ __attribute__((aligned(16))) __bf16 As[2][128][32];
  __shared__ __attribute__((aligned(16))) __bf16 Bs[2][128][32];

  const int tid  = threadIdx.x;
  const int lane = tid & 31;
  const int wave = tid >> 5;
  const int wm = wave >> 2;    // 0..1
  const int wn = wave & 3;     // 0..3
  const int blockM = blockIdx.y * 128;
  const int blockN = blockIdx.x * 128;

  // staging: each thread moves 32B of A-tile and 32B of B-tile per K step
  const int srow = tid >> 1;            // 0..127
  const int scol = (tid & 1) * 16;      // element offset 0 / 16

  v8f acc[4][2];
#pragma unroll
  for (int i = 0; i < 4; i++)
#pragma unroll
    for (int j = 0; j < 2; j++)
#pragma unroll
      for (int e = 0; e < 8; e++) acc[i][j][e] = 0.f;

  const int lrow = lane & 15;
  const int lhi  = (lane >> 4) * 8;

  const __bf16* gA = X + (size_t)(blockM + srow) * K + scol;
  const __bf16* gB = W + (size_t)(blockN + srow) * K + scol;
  const uint32_t la[2] = { lds_off(&As[0][srow][scol]), lds_off(&As[1][srow][scol]) };
  const uint32_t lb[2] = { lds_off(&Bs[0][srow][scol]), lds_off(&Bs[1][srow][scol]) };

  const int nk = K >> 5;

  // prologue: stage tile 0 into buffer 0
  async_cp16(la[0], gA);      async_cp16_off16(la[0], gA);
  async_cp16(lb[0], gB);      async_cp16_off16(lb[0], gB);

  for (int t = 0; t < nk; ++t) {
    const int cur = t & 1;
    if (t + 1 < nk) {
      // issue next tile into the other buffer (its readers were fenced by the
      // previous end-of-iteration barrier), then wait for the oldest 4 copies
      const __bf16* a = gA + (size_t)(t + 1) * 32;
      const __bf16* b = gB + (size_t)(t + 1) * 32;
      async_cp16(la[cur ^ 1], a); async_cp16_off16(la[cur ^ 1], a);
      async_cp16(lb[cur ^ 1], b); async_cp16_off16(lb[cur ^ 1], b);
      asm volatile("s_wait_asynccnt 0x4" ::: "memory");
    } else {
      asm volatile("s_wait_asynccnt 0x0" ::: "memory");
    }
    __syncthreads();   // all waves' tile-t data visible in LDS

    v16bf af[4], bf[2];
#pragma unroll
    for (int mi = 0; mi < 4; mi++) {
      const __bf16* p = &As[cur][wm * 64 + mi * 16 + lrow][lhi];
      ((v8bf*)&af[mi])[0] = *(const v8bf*)p;
      ((v8bf*)&af[mi])[1] = *(const v8bf*)(p + 16);
    }
#pragma unroll
    for (int ni = 0; ni < 2; ni++) {
      const __bf16* p = &Bs[cur][wn * 32 + ni * 16 + lrow][lhi];
      ((v8bf*)&bf[ni])[0] = *(const v8bf*)p;
      ((v8bf*)&bf[ni])[1] = *(const v8bf*)(p + 16);
    }
#pragma unroll
    for (int mi = 0; mi < 4; mi++)
#pragma unroll
      for (int ni = 0; ni < 2; ni++)
        acc[mi][ni] = __builtin_amdgcn_wmma_f32_16x16x32_bf16(
            false, af[mi], false, bf[ni], (short)0, acc[mi][ni], false, false);

    __syncthreads();   // buffer `cur` free for the stage issued at iter t+1
  }

  // epilogue (C layout: lane%16 -> col, (lane>>4)*8 + vgpr -> row)
#pragma unroll
  for (int ni = 0; ni < 2; ni++) {
    const int col = blockN + wn * 32 + ni * 16 + lrow;
    const float bv = bias ? bias[col] : 0.f;
#pragma unroll
    for (int mi = 0; mi < 4; mi++) {
#pragma unroll
      for (int r = 0; r < 8; r++) {
        const int row = blockM + wm * 64 + mi * 16 + (lane >> 4) * 8 + r;
        float v = acc[mi][ni][r] + bv;
        if (ACT == 1) v = (v > 0.f) ? v : (expf(v) - 1.f);
        if (Cf) Cf[(size_t)row * N + col] = v;
        if (Cb) Cb[(size_t)row * N + col] = (__bf16)v;
      }
    }
  }
}

// ---------------------------------------------------------------------------
// Launch
// ---------------------------------------------------------------------------

extern "C" void kernel_launch(void* const* d_in, const int* in_sizes, int n_in,
                              void* d_out, int out_size, void* d_ws, size_t ws_size,
                              hipStream_t stream) {
  (void)in_sizes; (void)n_in; (void)out_size; (void)ws_size;

  const float* action = (const float*)d_in[0];
  const float* prev_s = (const float*)d_in[1];
  const float* prev_h = (const float*)d_in[2];
  const float* eps    = (const float*)d_in[3];
  const float* W_in   = (const float*)d_in[4];
  const float* b_in   = (const float*)d_in[5];
  const float* W_ih   = (const float*)d_in[6];
  const float* W_hh   = (const float*)d_in[7];
  const float* b_ih   = (const float*)d_in[8];
  const float* b_n    = (const float*)d_in[9];
  const float* W_h    = (const float*)d_in[10];
  const float* b_h    = (const float*)d_in[11];
  const float* W_out  = (const float*)d_in[12];
  const float* b_out  = (const float*)d_in[13];

  float* out  = (float*)d_out;
  float* mean = out;
  float* stdv = out + (size_t)NB * NS;
  float* samp = out + (size_t)2 * NB * NS;
  float* nh_f = out + (size_t)3 * NB * NS;   // new_rnn_hidden, [NB,NR]

  // workspace bump allocator
  char* p = (char*)d_ws;
  auto alloc = [&](size_t bytes) -> void* {
    void* r = (void*)p;
    p += (bytes + 255) & ~(size_t)255;
    return r;
  };
  __bf16* xcat   = (__bf16*)alloc((size_t)NB * NK0 * 2);
  __bf16* winb   = (__bf16*)alloc((size_t)NH * NK0 * 2);
  __bf16* wihb   = (__bf16*)alloc((size_t)3 * NR * NH * 2);
  __bf16* whhb   = (__bf16*)alloc((size_t)3 * NR * NR * 2);
  __bf16* whb    = (__bf16*)alloc((size_t)NH * NR * 2);
  __bf16* woutb  = (__bf16*)alloc((size_t)2 * NS * NH * 2);
  __bf16* prevhb = (__bf16*)alloc((size_t)NB * NR * 2);
  __bf16* hb     = (__bf16*)alloc((size_t)NB * NH * 2);
  __bf16* nhb    = (__bf16*)alloc((size_t)NB * NR * 2);
  __bf16* h2b    = (__bf16*)alloc((size_t)NB * NH * 2);
  float*  igates = (float*)alloc((size_t)NB * 3 * NR * 4);
  float*  hgates = (float*)alloc((size_t)NB * 3 * NR * 4);
  float*  outf   = igates;  // alias: igates dead after GRU, GEMM5 output reuses it

  auto grid1d = [](size_t n) { return dim3((unsigned)((n + 255) / 256)); };

  // 1) pack + convert to bf16
  pack_xcat<<<grid1d((size_t)NB * NK0), 256, 0, stream>>>(action, prev_s, xcat);
  cvt_f32_bf16x4<<<grid1d((size_t)NH * NK0 / 4), 256, 0, stream>>>(W_in, winb, NH * NK0 / 4);
  cvt_f32_bf16x4<<<grid1d((size_t)3 * NR * NH / 4), 256, 0, stream>>>(W_ih, wihb, 3 * NR * NH / 4);
  cvt_f32_bf16x4<<<grid1d((size_t)3 * NR * NR / 4), 256, 0, stream>>>(W_hh, whhb, 3 * NR * NR / 4);
  cvt_f32_bf16x4<<<grid1d((size_t)NH * NR / 4), 256, 0, stream>>>(W_h, whb, NH * NR / 4);
  cvt_f32_bf16x4<<<grid1d((size_t)2 * NS * NH / 4), 256, 0, stream>>>(W_out, woutb, 2 * NS * NH / 4);
  cvt_f32_bf16x4<<<grid1d((size_t)NB * NR / 4), 256, 0, stream>>>(prev_h, prevhb, NB * NR / 4);

  // 2) h = elu(xcat @ W_in^T + b_in)           [NB,NH]  (bf16 for next GEMM)
  gemm_bt<1><<<dim3(NH / 128, NB / 128), 256, 0, stream>>>(
      xcat, winb, b_in, (float*)nullptr, hb, NB, NH, NK0);

  // 3) igates = h @ W_ih^T + b_ih              [NB,3R]  (f32)
  gemm_bt<0><<<dim3(3 * NR / 128, NB / 128), 256, 0, stream>>>(
      hb, wihb, b_ih, igates, (__bf16*)nullptr, NB, 3 * NR, NH);

  // 4) hgates = prev_h @ W_hh^T                [NB,3R]  (f32)
  gemm_bt<0><<<dim3(3 * NR / 128, NB / 128), 256, 0, stream>>>(
      prevhb, whhb, (const float*)nullptr, hgates, (__bf16*)nullptr, NB, 3 * NR, NR);

  // 5) GRU gate math -> new hidden (f32 into d_out, bf16 for GEMM)
  gru_eltwise<<<grid1d((size_t)NB * NR), 256, 0, stream>>>(
      igates, hgates, prev_h, b_n, nh_f, nhb);

  // 6) h2 = elu(new_h @ W_h^T + b_h)           [NB,NH]  (bf16)
  gemm_bt<1><<<dim3(NH / 128, NB / 128), 256, 0, stream>>>(
      nhb, whb, b_h, (float*)nullptr, h2b, NB, NH, NR);

  // 7) out = h2 @ W_out^T + b_out              [NB,2S]  (f32)
  gemm_bt<0><<<dim3(2 * NS / 128, NB / 128), 256, 0, stream>>>(
      h2b, woutb, b_out, outf, (__bf16*)nullptr, NB, 2 * NS, NH);

  // 8) mean / std / sample
  head_eltwise<<<grid1d((size_t)NB * NS), 256, 0, stream>>>(outf, eps, mean, stdv, samp);
}
